// GINConv_ptens_50869592655547
// MI455X (gfx1250) — matrix operand order repaired and checked by reading, not compiled, not run on hardware
//
#include <hip/hip_runtime.h>
#include <hip/hip_bf16.h>

typedef __attribute__((ext_vector_type(16))) __bf16        v16bf;
typedef __attribute__((ext_vector_type(8)))  float         v8f;
typedef __attribute__((ext_vector_type(4)))  unsigned int  v4u;
typedef __attribute__((ext_vector_type(8)))  int           v8i;
typedef __attribute__((ext_vector_type(4)))  int           v4i;

#define D_IN  128
#define D_HID 512
#define WAVES_PER_BLOCK 16
#define ROWS_PER_WAVE   16
#define ROWS_PER_BLOCK  (WAVES_PER_BLOCK * ROWS_PER_WAVE)   // 256

#define W1_HALFS (D_IN * D_HID)        // 65536
#define W2_HALFS (D_HID * D_IN)        // 65536
#define PACKED_HALFS (W1_HALFS + W2_HALFS)   // 131072 -> 256 KB bf16

#if __has_builtin(__builtin_amdgcn_tensor_load_to_lds)
#define HAVE_TDM 1
#else
#define HAVE_TDM 0
#endif

// ---------------------------------------------------------------- zero agg
__global__ void gin_zero_kernel(float* __restrict__ p, int n4) {
    float4 z = {0.f, 0.f, 0.f, 0.f};
    for (int i = blockIdx.x * blockDim.x + threadIdx.x; i < n4;
         i += gridDim.x * blockDim.x)
        ((float4*)p)[i] = z;
}

// ------------------------------------------------- edge scatter (atomics)
// One wave per edge: lane l owns float4 chunk l of the 128-float row.
// agg[src] += x[dst]; agg[dst] += x[src]  (== neighbor sum; the reference's
// deg*x correction cancels algebraically).
__global__ __launch_bounds__(256) void gin_scatter_kernel(
    const float* __restrict__ x, const int* __restrict__ ei,
    float* __restrict__ agg, int E)
{
    int edge = blockIdx.x * 8 + (threadIdx.x >> 5);
    if (edge >= E) return;
    int l = threadIdx.x & 31;
    int s = ei[edge];
    int d = ei[E + edge];
    float4 vs = ((const float4*)(x + (size_t)s * D_IN))[l];
    float4 vd = ((const float4*)(x + (size_t)d * D_IN))[l];
    float* as_ = agg + (size_t)s * D_IN + l * 4;
    float* ad_ = agg + (size_t)d * D_IN + l * 4;
    atomicAdd(as_ + 0, vd.x); atomicAdd(as_ + 1, vd.y);
    atomicAdd(as_ + 2, vd.z); atomicAdd(as_ + 3, vd.w);
    atomicAdd(ad_ + 0, vs.x); atomicAdd(ad_ + 1, vs.y);
    atomicAdd(ad_ + 2, vs.z); atomicAdd(ad_ + 3, vs.w);
}

// ------------------------------------------- one-time weight pack (f32->bf16)
// Writes W1 then W2 into d_ws in WMMA-B-operand tile order:
//   tile index [nt][ks][lane][i];  lane l holds column n = l&15,
//   K element (l>>4)*16 + i within a K=32 x N=16 tile.
__global__ __launch_bounds__(512) void gin_pack_weights(
    const float* __restrict__ W1, const float* __restrict__ W2,
    __bf16* __restrict__ pw)
{
    int idx = blockIdx.x * blockDim.x + threadIdx.x;
    if (idx < W1_HALFS) {
        int k = idx >> 9, n = idx & 511;               // W1[k][n]
        int nt = n >> 4, nl = n & 15, ks = k >> 5, kw = k & 31;
        int lane = ((kw >> 4) << 4) + nl, i = kw & 15;
        pw[(((nt * 4 + ks) * 32) + lane) * 16 + i] = (__bf16)W1[idx];
    } else if (idx < PACKED_HALFS) {
        int j = idx - W1_HALFS;
        int k = j >> 7, n = j & 127;                   // W2[k][n]
        int nt = n >> 4, nl = n & 15, ks = k >> 5, kw = k & 31;
        int lane = ((kw >> 4) << 4) + nl, i = kw & 15;
        pw[W1_HALFS + (((nt * 16 + ks) * 32) + lane) * 16 + i] = (__bf16)W2[j];
    }
}

// ------------------------------------------------------- fused WMMA MLP
// LDS: 256 KB packed bf16 weights (TDM bulk load) + 32 KB per-wave staging
// for the C->A cross-lane relayout = 288 KB (within 320 KB WGP budget).
__global__ __launch_bounds__(32 * WAVES_PER_BLOCK) void gin_mlp_kernel(
    const float* __restrict__ x, const float* __restrict__ agg,
    const float* __restrict__ eps,
    const float* __restrict__ W1, const float* __restrict__ b1,
    const float* __restrict__ W2, const float* __restrict__ b2,
    const __bf16* __restrict__ packedW,     // may be null -> in-kernel convert
    float* __restrict__ out, int nrows)
{
    __shared__ __align__(16) __bf16 sW[PACKED_HALFS];
    __shared__ __align__(16) float  sStage[WAVES_PER_BLOCK][16 * 32];
    __bf16* sW1 = sW;
    __bf16* sW2 = sW + W1_HALFS;

    const int tid = threadIdx.x;
    const int nthreads = blockDim.x;

    if (packedW) {
#if HAVE_TDM
        // --- CDNA5 Tensor Data Mover: one 256 KB DMA, global -> LDS -------
        if (tid < 32) {
            unsigned int       ldsa = (unsigned int)(uintptr_t)(void*)sW;
            unsigned long long ga   = (unsigned long long)(uintptr_t)packedW;
            const unsigned int elems = (PACKED_HALFS * 2u) / 8u;  // 8B units

            v4u g0;
            g0.x = 1u;                                   // count=1, user mode
            g0.y = ldsa;                                 // lds_addr
            g0.z = (unsigned int)ga;                     // global_addr[31:0]
            g0.w = (unsigned int)((ga >> 32) & 0x01FFFFFFu) | 0x80000000u; // addr[56:32] | type=2

            v8i g1;
            g1[0] = (int)(3u << 16);                     // data_size=3 (8B)
            g1[1] = (int)((elems & 0xFFFFu) << 16);      // tensor_dim0[15:0]
            g1[2] = (int)((elems >> 16) | (1u << 16));   // tensor_dim0[31:16], tensor_dim1=1
            g1[3] = (int)((elems & 0xFFFFu) << 16);      // tile_dim0 = elems
            g1[4] = 1;                                   // tile_dim1=1, tile_dim2=0
            g1[5] = (int)elems;                          // tensor_dim0_stride
            g1[6] = 0;
            g1[7] = 0;

            v4i gz = (v4i){};
#if defined(__clang_major__) && __clang_major__ >= 23
            v8i gz8 = (v8i){};
            __builtin_amdgcn_tensor_load_to_lds(g0, g1, gz, gz, gz8, 0);
#else
            __builtin_amdgcn_tensor_load_to_lds(g0, g1, gz, gz, 0);
#endif
            __builtin_amdgcn_s_wait_tensorcnt(0);
        }
#else
        // --- fallback: plain vectorized bulk copy into LDS ---------------
        const float4* src = (const float4*)packedW;      // 16384 x 16B
        float4* dst = (float4*)sW;
        for (int c = tid; c < (PACKED_HALFS * 2) / 16; c += nthreads)
            dst[c] = src[c];
#endif
    } else {
        // --- workspace unavailable: convert + swizzle in-kernel ----------
        for (int idx = tid; idx < W1_HALFS; idx += nthreads) {
            int k = idx >> 9, n = idx & 511;
            int nt = n >> 4, nl = n & 15, ks = k >> 5, kw = k & 31;
            int lane = ((kw >> 4) << 4) + nl, i = kw & 15;
            sW1[(((nt * 4 + ks) * 32) + lane) * 16 + i] = (__bf16)W1[idx];
        }
        for (int idx = tid; idx < W2_HALFS; idx += nthreads) {
            int k = idx >> 7, n = idx & 127;
            int nt = n >> 4, nl = n & 15, ks = k >> 5, kw = k & 31;
            int lane = ((kw >> 4) << 4) + nl, i = kw & 15;
            sW2[(((nt * 16 + ks) * 32) + lane) * 16 + i] = (__bf16)W2[idx];
        }
    }
    __syncthreads();

    const int wave = tid >> 5;
    const int l    = tid & 31;
    const int nlo  = l & 15;
    const int hi   = l >> 4;
    const float scale = 1.0f + eps[0];

    const int rowBase = blockIdx.x * ROWS_PER_BLOCK + wave * ROWS_PER_WAVE;
    int r  = rowBase + nlo;                      // this lane's A-matrix row
    int rc = (r < nrows) ? r : (nrows - 1);      // clamp for tail tile

    // ---- build A = bf16((1+eps)*x + agg), 16x128, 4 K-slices of 32 ----
    v16bf A[4];
    const float4* xr = (const float4*)(x   + (size_t)rc * D_IN);
    const float4* gr = (const float4*)(agg + (size_t)rc * D_IN);
#pragma unroll
    for (int kk = 0; kk < 4; ++kk) {
        int q0 = (kk * 32 + hi * 16) >> 2;       // elems 0..7  : K = kk*32 + hi*16 + i
        int q1 = q0 + 2;                         // elems 8..15 : K = +8
        float4 xa = xr[q0], xb = xr[q0 + 1], xc = xr[q1], xd = xr[q1 + 1];
        float4 ga = gr[q0], gb = gr[q0 + 1], gc = gr[q1], gd = gr[q1 + 1];
        float fx[16] = {xa.x, xa.y, xa.z, xa.w, xb.x, xb.y, xb.z, xb.w,
                        xc.x, xc.y, xc.z, xc.w, xd.x, xd.y, xd.z, xd.w};
        float fg[16] = {ga.x, ga.y, ga.z, ga.w, gb.x, gb.y, gb.z, gb.w,
                        gc.x, gc.y, gc.z, gc.w, gd.x, gd.y, gd.z, gd.w};
        v16bf a;
#pragma unroll
        for (int i = 0; i < 16; ++i) a[i] = (__bf16)(scale * fx[i] + fg[i]);
        A[kk] = a;
    }

    v8f acc2[8];
#pragma unroll
    for (int nt = 0; nt < 8; ++nt) acc2[nt] = (v8f){};

    float* st = sStage[wave];

    // ---- GEMM1 (relu) fused into GEMM2, 32 hidden columns at a time ----
    for (int hb = 0; hb < 16; ++hb) {
#pragma unroll
        for (int t = 0; t < 2; ++t) {
            int ntile = hb * 2 + t;
            v8f c = (v8f){};
#pragma unroll
            for (int ks = 0; ks < 4; ++ks) {
                const v16bf b =
                    *(const v16bf*)&sW1[(((ntile * 4 + ks) * 32) + l) * 16];
                c = __builtin_amdgcn_wmma_f32_16x16x32_bf16(
                        false, A[ks], false, b, (short)0, c, false, false);
            }
            float bias = b1[ntile * 16 + nlo];
#pragma unroll
            for (int rr = 0; rr < 8; ++rr) {
                float h = c[rr] + bias;
                h = h > 0.f ? h : 0.f;           // relu
                st[(hi * 8 + rr) * 32 + t * 16 + nlo] = h;
            }
        }
        // cross-lane relayout through LDS: wait for this wave's DS stores
        asm volatile("s_wait_dscnt 0" ::: "memory");

        v16bf a2;
        {
            const float4* sp = (const float4*)&st[nlo * 32 + hi * 16];
            const float4* sq = (const float4*)&st[nlo * 32 + hi * 16 + 8];
            float4 f0 = sp[0], f1 = sp[1], f2 = sq[0], f3 = sq[1];
            float fl[16] = {f0.x, f0.y, f0.z, f0.w, f1.x, f1.y, f1.z, f1.w,
                            f2.x, f2.y, f2.z, f2.w, f3.x, f3.y, f3.z, f3.w};
#pragma unroll
            for (int i = 0; i < 16; ++i) a2[i] = (__bf16)fl[i];
        }
#pragma unroll
        for (int nt = 0; nt < 8; ++nt) {
            const v16bf b =
                *(const v16bf*)&sW2[(((nt * 16 + hb) * 32) + l) * 16];
            acc2[nt] = __builtin_amdgcn_wmma_f32_16x16x32_bf16(
                           false, a2, false, b, (short)0, acc2[nt], false, false);
        }
    }

    // ---- epilogue: + b2, write 16x128 f32 tile ----
#pragma unroll
    for (int nt = 0; nt < 8; ++nt) {
        float bias = b2[nt * 16 + nlo];
#pragma unroll
        for (int rr = 0; rr < 8; ++rr) {
            int row = rowBase + hi * 8 + rr;
            if (row < nrows)
                out[(size_t)row * D_IN + nt * 16 + nlo] = acc2[nt][rr] + bias;
        }
    }
}

// ------------------------------------------------------------------ launch
extern "C" void kernel_launch(void* const* d_in, const int* in_sizes, int n_in,
                              void* d_out, int out_size, void* d_ws, size_t ws_size,
                              hipStream_t stream) {
    const float* x   = (const float*)d_in[0];
    const int*   ei  = (const int*)  d_in[1];
    const float* eps = (const float*)d_in[2];
    const float* W1  = (const float*)d_in[3];
    const float* b1  = (const float*)d_in[4];
    const float* W2  = (const float*)d_in[5];
    const float* b2  = (const float*)d_in[6];
    float* out = (float*)d_out;

    const int N = in_sizes[0] / D_IN;   // 50000
    const int E = in_sizes[1] / 2;      // 800000

    // d_out doubles as the f32 neighbor-sum accumulator: zeroed every call,
    // scatter-added, then overwritten tile-by-tile by the fused MLP (each
    // block reads only its own rows before writing them).
    float* agg = out;

    gin_zero_kernel<<<2048, 256, 0, stream>>>(agg, (N * D_IN) / 4);
    gin_scatter_kernel<<<(E + 7) / 8, 256, 0, stream>>>(x, ei, agg, E);

    // Pack weights once per call into d_ws (bf16, WMMA tile order) if the
    // workspace is big enough; otherwise the MLP kernel converts in-LDS.
    bool use_ws = ws_size >= (size_t)(PACKED_HALFS * sizeof(__bf16));
    __bf16* pw = use_ws ? (__bf16*)d_ws : (__bf16*)nullptr;
    if (use_ws)
        gin_pack_weights<<<PACKED_HALFS / 512, 512, 0, stream>>>(W1, W2, pw);

    int blocks = (N + ROWS_PER_BLOCK - 1) / ROWS_PER_BLOCK;
    gin_mlp_kernel<<<blocks, 32 * WAVES_PER_BLOCK, 0, stream>>>(
        x, agg, eps, W1, b1, W2, b2, pw, out, N);
}
